// BiLSTM_CRF_77902116815269
// MI455X (gfx1250) — matrix operand order repaired and checked by reading, not compiled, not run on hardware
//
#include <hip/hip_runtime.h>
#include <hip/hip_bf16.h>
#include <math.h>

// ---------------------------------------------------------------------------
// BiLSTM-CRF for MI455X (gfx1250), wave32 + v_wmma_f32_16x16x32_bf16.
//   time steps BT=32 (reference batch dim), lstm batch SB=512, E=256, H=256.
// ---------------------------------------------------------------------------

typedef __attribute__((ext_vector_type(16))) __bf16 v16bf;
typedef __attribute__((ext_vector_type(8)))  float  v8f;

#define BT   32      // scan/time steps (reference "B")
#define SB   512     // lstm batch       (reference "S")
#define EDIM 256
#define HDIM 256
#define G4   1024    // 4*H gates
#define KCAT 512     // E + H fused K
#define HD2  512     // concat hidden (2*H)
#define NT   11
#define NTP  16      // padded tag count
#define NEGV -10000.0f

__device__ __forceinline__ __bf16 bfc(float f) {
  union { float f; unsigned u; } x; x.f = f;
  unsigned r = x.u + 0x7fffu + ((x.u >> 16) & 1u);   // round-to-nearest-even
  unsigned short h = (unsigned short)(r >> 16);
  union { unsigned short s; __bf16 b; } y; y.s = h;
  return y.b;
}

__device__ __forceinline__ float sigf(float x) { return 1.0f / (1.0f + expf(-x)); }

// A fragment (16x32 bf16, row-major source, stride ld elements).
// lane: row = base row + (lane&15); k0 already includes (lane>=16 ? 8 : 0).
__device__ __forceinline__ v16bf load_a(const __bf16* __restrict__ base, int ld,
                                        int row, int k0) {
  v16bf f;
  const __bf16* p = base + (size_t)row * ld + k0;
#pragma unroll
  for (int i = 0; i < 8; ++i) f[i] = p[i];
#pragma unroll
  for (int i = 0; i < 8; ++i) f[8 + i] = p[16 + i];
  return f;
}

// B fragment (32x16 bf16). B[k][n] = W[n][k] with W row-major stride ld.
// lane: n = base n + (lane&15); k0 already includes (lane>=16 ? 8 : 0).
__device__ __forceinline__ v16bf load_b(const __bf16* __restrict__ base, int ld,
                                        int row, int k0) {
  v16bf f;
  const __bf16* p = base + (size_t)row * ld + k0;
#pragma unroll
  for (int i = 0; i < 8; ++i) f[i] = p[i];
#pragma unroll
  for (int i = 0; i < 8; ++i) f[8 + i] = p[16 + i];
  return f;
}

// ---------------------------------------------------------------------------
// Prep kernels
// ---------------------------------------------------------------------------

__global__ void k_embed(const int* __restrict__ sent, const float* __restrict__ embed,
                        __bf16* __restrict__ xb) {
  const size_t total = (size_t)BT * SB * EDIM;
  for (size_t i = (size_t)blockIdx.x * blockDim.x + threadIdx.x; i < total;
       i += (size_t)gridDim.x * blockDim.x) {
    size_t row = i >> 8;            // / EDIM
    int    e   = (int)(i & 255);
    int    v   = sent[row];
    xb[i] = bfc(embed[(size_t)v * EDIM + e]);
  }
}

__global__ void k_wcat(const float* __restrict__ wih, const float* __restrict__ whh,
                       __bf16* __restrict__ wcat) {
  const size_t total = (size_t)G4 * KCAT;
  for (size_t i = (size_t)blockIdx.x * blockDim.x + threadIdx.x; i < total;
       i += (size_t)gridDim.x * blockDim.x) {
    int n = (int)(i >> 9);
    int k = (int)(i & 511);
    float v = (k < HDIM) ? wih[(size_t)n * EDIM + k]
                         : whh[(size_t)n * HDIM + (k - EDIM)];
    wcat[i] = bfc(v);
  }
}

__global__ void k_bias(const float* __restrict__ bih, const float* __restrict__ bhh,
                       float* __restrict__ bias) {
  int i = blockIdx.x * blockDim.x + threadIdx.x;
  if (i < G4) bias[i] = bih[i] + bhh[i];
}

__global__ void k_h0c0(const float* __restrict__ h0, const float* __restrict__ c0,
                       __bf16* __restrict__ h0b, float* __restrict__ cbuf) {
  const size_t total = (size_t)2 * SB * HDIM;
  for (size_t i = (size_t)blockIdx.x * blockDim.x + threadIdx.x; i < total;
       i += (size_t)gridDim.x * blockDim.x) {
    h0b[i]  = bfc(h0[i]);
    cbuf[i] = c0[i];
  }
}

__global__ void k_wout(const float* __restrict__ wout, const float* __restrict__ bout,
                       __bf16* __restrict__ woutb, float* __restrict__ boutp) {
  const size_t total = (size_t)NTP * HD2;
  for (size_t i = (size_t)blockIdx.x * blockDim.x + threadIdx.x; i < total;
       i += (size_t)gridDim.x * blockDim.x) {
    int n = (int)(i >> 9);
    int k = (int)(i & 511);
    woutb[i] = bfc(n < NT ? wout[(size_t)n * HD2 + k] : 0.0f);
    if (i < NTP) boutp[i] = (i < NT) ? bout[i] : 0.0f;
  }
}

// ---------------------------------------------------------------------------
// Fused recurrent step: G = [x_t | h_{t-1}] @ [Wih|Whh]^T + bias, both dirs.
// Grid: 128 blocks x 256 threads = 1024 waves; blocks [0,64) fwd, [64,128) bwd.
// Each wave owns one 16-row x 16-hidden-col tile and computes all 4 gates.
// ---------------------------------------------------------------------------

__global__ void __launch_bounds__(256)
k_lstm_step(const __bf16* __restrict__ xb, __bf16* __restrict__ hs,
            const __bf16* __restrict__ wcatF, const __bf16* __restrict__ wcatB,
            const float* __restrict__ biasF, const float* __restrict__ biasB,
            const __bf16* __restrict__ h0b, float* __restrict__ cbuf, int j) {
  const int l     = threadIdx.x & 31;
  const int dir   = blockIdx.x >> 6;                       // 0 fwd, 1 bwd
  const int w     = (blockIdx.x & 63) * 8 + (threadIdx.x >> 5);  // 0..511
  const int mtile = w >> 4;                                // 0..31 (rows of SB)
  const int ntile = w & 15;                                // 0..15 (hidden cols)

  const int t_in = dir ? (BT - 1 - j) : j;
  const __bf16* wcat = dir ? wcatB : wcatF;
  const float*  bias = dir ? biasB : biasF;

  const __bf16* hprev;
  int hstride;
  if (j == 0) {
    hprev = h0b + (size_t)dir * SB * HDIM;
    hstride = HDIM;
  } else {
    const int tprev = dir ? (BT - j) : (j - 1);
    hprev = hs + (size_t)tprev * SB * HD2 + (size_t)dir * HDIM;
    hstride = HD2;
  }
  const __bf16* xrow = xb + (size_t)t_in * SB * EDIM;

  const int n16 = l & 15;
  const int hi  = l >> 4;
  const int aK  = hi * 8;
  const int bK  = hi * 8;
  const int arow  = mtile * 16 + n16;           // A row (s index)
  const int brow0 = ntile * 16 + n16;           // B row within gate block

  v8f acc0 = {}, acc1 = {}, acc2 = {}, acc3 = {};
#pragma unroll
  for (int ch = 0; ch < 16; ++ch) {
    const int kc = ch * 32;
    v16bf a = (ch < 8) ? load_a(xrow, EDIM, arow, kc + aK)
                       : load_a(hprev, hstride, arow, (kc - EDIM) + aK);
    v16bf b0 = load_b(wcat, KCAT, 0 * HDIM + brow0, kc + bK);
    v16bf b1 = load_b(wcat, KCAT, 1 * HDIM + brow0, kc + bK);
    v16bf b2 = load_b(wcat, KCAT, 2 * HDIM + brow0, kc + bK);
    v16bf b3 = load_b(wcat, KCAT, 3 * HDIM + brow0, kc + bK);
    acc0 = __builtin_amdgcn_wmma_f32_16x16x32_bf16(false, a, false, b0, (short)0, acc0, false, false);
    acc1 = __builtin_amdgcn_wmma_f32_16x16x32_bf16(false, a, false, b1, (short)0, acc1, false, false);
    acc2 = __builtin_amdgcn_wmma_f32_16x16x32_bf16(false, a, false, b2, (short)0, acc2, false, false);
    acc3 = __builtin_amdgcn_wmma_f32_16x16x32_bf16(false, a, false, b3, (short)0, acc3, false, false);
  }

  // LSTM nonlinearity fused in-register (PyTorch gate order i,f,g,o).
  const int   hc  = ntile * 16 + n16;
  const float bi  = bias[hc];
  const float bff = bias[HDIM + hc];
  const float bg  = bias[2 * HDIM + hc];
  const float bo  = bias[3 * HDIM + hc];
  float*  crow = cbuf + (size_t)dir * SB * HDIM;
  __bf16* hout = hs + (size_t)t_in * SB * HD2 + (size_t)dir * HDIM;
#pragma unroll
  for (int v = 0; v < 8; ++v) {
    const int s  = mtile * 16 + hi * 8 + v;
    const float gi = acc0[v] + bi;
    const float gf = acc1[v] + bff;
    const float gg = acc2[v] + bg;
    const float go = acc3[v] + bo;
    const float c  = crow[(size_t)s * HDIM + hc];
    const float cn = sigf(gf) * c + sigf(gi) * tanhf(gg);
    crow[(size_t)s * HDIM + hc] = cn;
    hout[(size_t)s * HD2 + hc]  = bfc(sigf(go) * tanhf(cn));
  }
}

// ---------------------------------------------------------------------------
// Output projection: feats[r][0..15] = hs[r][:]@Wout^T + bout  (stride NTP)
// 1024 waves, one 16x16 tile each, K = 512.
// ---------------------------------------------------------------------------

__global__ void __launch_bounds__(256)
k_proj(const __bf16* __restrict__ hs, const __bf16* __restrict__ woutb,
       const float* __restrict__ boutp, float* __restrict__ feats) {
  const int l   = threadIdx.x & 31;
  const int w   = blockIdx.x * 8 + (threadIdx.x >> 5);  // 0..1023
  const int r0  = w * 16;
  const int n16 = l & 15;
  const int hi  = l >> 4;

  v8f acc = {};
#pragma unroll
  for (int ch = 0; ch < 16; ++ch) {
    v16bf a = load_a(hs, HD2, r0 + n16, ch * 32 + hi * 8);
    v16bf b = load_b(woutb, HD2, n16, ch * 32 + hi * 8);
    acc = __builtin_amdgcn_wmma_f32_16x16x32_bf16(false, a, false, b, (short)0, acc, false, false);
  }
  const float bb = boutp[n16];
#pragma unroll
  for (int v = 0; v < 8; ++v) {
    feats[(size_t)(r0 + hi * 8 + v) * NTP + n16] = acc[v] + bb;
  }
}

// ---------------------------------------------------------------------------
// Viterbi: one wave32, lane = batch (32 batches), time = 512, tags = 11.
// ---------------------------------------------------------------------------

__global__ void k_viterbi(const float* __restrict__ feats, const float* __restrict__ trans,
                          unsigned char* __restrict__ psi, float* __restrict__ out) {
  __shared__ float tr[NT * NT];
  const int l = threadIdx.x;
  for (int i = l; i < NT * NT; i += 32) tr[i] = trans[i];
  __syncthreads();

  const int b = l;
  float d[NT];
#pragma unroll
  for (int i = 0; i < NT; ++i) d[i] = NEGV;
  d[9] = 0.0f;  // START tag

  const float* fb = feats + (size_t)b * SB * NTP;
  for (int t = 1; t < SB; ++t) {
    float nd[NT];
    unsigned char* pp = psi + ((size_t)(t - 1) * BT + b) * NTP;
#pragma unroll
    for (int to = 0; to < NT; ++to) {
      float best = tr[to * NT + 0] + d[0];
      int   arg  = 0;
#pragma unroll
      for (int fr = 1; fr < NT; ++fr) {
        const float v = tr[to * NT + fr] + d[fr];
        if (v > best) { best = v; arg = fr; }   // first-max tie-break (argmax)
      }
      nd[to] = best + fb[(size_t)t * NTP + to];
      pp[to] = (unsigned char)arg;
    }
#pragma unroll
    for (int i = 0; i < NT; ++i) d[i] = nd[i];
  }

  float best = d[0];
  int   last = 0;
#pragma unroll
  for (int i = 1; i < NT; ++i)
    if (d[i] > best) { best = d[i]; last = i; }

  out[b] = best;                                 // score (B,)
  float* path = out + BT;                        // path (B, S) as floats
  path[(size_t)b * SB + (SB - 1)] = (float)last;
  int tag = last;
  for (int t = SB - 2; t >= 0; --t) {
    tag = psi[((size_t)t * BT + b) * NTP + tag];
    path[(size_t)b * SB + t] = (float)tag;
  }
}

// ---------------------------------------------------------------------------
// Host launcher
// ---------------------------------------------------------------------------

extern "C" void kernel_launch(void* const* d_in, const int* in_sizes, int n_in,
                              void* d_out, int out_size, void* d_ws, size_t ws_size,
                              hipStream_t stream) {
  (void)in_sizes; (void)n_in; (void)out_size; (void)ws_size;

  const int*   sentence = (const int*)d_in[0];
  const float* embed    = (const float*)d_in[1];
  const float* w_ih_f   = (const float*)d_in[2];
  const float* w_hh_f   = (const float*)d_in[3];
  const float* b_ih_f   = (const float*)d_in[4];
  const float* b_hh_f   = (const float*)d_in[5];
  const float* w_ih_b   = (const float*)d_in[6];
  const float* w_hh_b   = (const float*)d_in[7];
  const float* b_ih_b   = (const float*)d_in[8];
  const float* b_hh_b   = (const float*)d_in[9];
  const float* h0       = (const float*)d_in[10];
  const float* c0       = (const float*)d_in[11];
  const float* w_out    = (const float*)d_in[12];
  const float* b_out    = (const float*)d_in[13];
  const float* transit  = (const float*)d_in[14];

  char* base = (char*)d_ws;
  size_t off = 0;
  auto take = [&](size_t bytes) -> char* {
    char* p = base + off;
    off += (bytes + 255) & ~(size_t)255;
    return p;
  };

  __bf16* xb    = (__bf16*)take((size_t)BT * SB * EDIM * 2);
  __bf16* hs    = (__bf16*)take((size_t)BT * SB * HD2 * 2);
  __bf16* wcatF = (__bf16*)take((size_t)G4 * KCAT * 2);
  __bf16* wcatB = (__bf16*)take((size_t)G4 * KCAT * 2);
  float*  biasF = (float*)take((size_t)G4 * 4);
  float*  biasB = (float*)take((size_t)G4 * 4);
  __bf16* h0b   = (__bf16*)take((size_t)2 * SB * HDIM * 2);
  float*  cbuf  = (float*)take((size_t)2 * SB * HDIM * 4);
  __bf16* woutb = (__bf16*)take((size_t)NTP * HD2 * 2);
  float*  boutp = (float*)take((size_t)NTP * 4);
  float*  feats = (float*)take((size_t)BT * SB * NTP * 4);
  unsigned char* psi = (unsigned char*)take((size_t)(SB - 1) * BT * NTP);

  // Prep
  k_embed<<<2048, 256, 0, stream>>>(sentence, embed, xb);
  k_wcat<<<512, 256, 0, stream>>>(w_ih_f, w_hh_f, wcatF);
  k_wcat<<<512, 256, 0, stream>>>(w_ih_b, w_hh_b, wcatB);
  k_bias<<<4, 256, 0, stream>>>(b_ih_f, b_hh_f, biasF);
  k_bias<<<4, 256, 0, stream>>>(b_ih_b, b_hh_b, biasB);
  k_h0c0<<<256, 256, 0, stream>>>(h0, c0, h0b, cbuf);
  k_wout<<<32, 256, 0, stream>>>(w_out, b_out, woutb, boutp);

  // Recurrence: 32 sequential fused WMMA steps (both directions per launch)
  for (int j = 0; j < BT; ++j) {
    k_lstm_step<<<128, 256, 0, stream>>>(xb, hs, wcatF, wcatB, biasF, biasB,
                                         h0b, cbuf, j);
  }

  // Projection + Viterbi decode
  k_proj<<<128, 256, 0, stream>>>(hs, woutb, boutp, feats);
  k_viterbi<<<1, 32, 0, stream>>>(feats, transit, psi, (float*)d_out);
}
